// MultiHeadAttention_66228395705035
// MI455X (gfx1250) — matrix operand order repaired and checked by reading, not compiled, not run on hardware
//
#include <hip/hip_runtime.h>

// MI455X / gfx1250: wave32, WMMA 16x16x32 bf16 (f32 accum), TDM async tile loads.
typedef __bf16 bf16_t;
typedef __attribute__((ext_vector_type(16))) __bf16 v16bf;
typedef __attribute__((ext_vector_type(8)))  __bf16 v8bf;
typedef __attribute__((ext_vector_type(8)))  float  v8f;

typedef unsigned int tdm_v4u __attribute__((ext_vector_type(4)));
typedef int          tdm_v8i __attribute__((ext_vector_type(8)));
typedef int          tdm_v4i __attribute__((ext_vector_type(4)));

#if defined(__has_builtin)
#if __has_builtin(__builtin_amdgcn_tensor_load_to_lds)
#define USE_TDM 1
#else
#define USE_TDM 0
#endif
#else
#define USE_TDM 0
#endif

#define B_  4
#define T_  2048
#define D_  1024
#define H_  16
#define K_  64
#define HK_ 1024
#define M_  8192          // B_*T_
#define AST 40            // LDS row stride for 32-wide rows (pad vs bank conflicts, 16B align)
#define KST 72            // LDS row stride for 64-wide rows

__device__ __forceinline__ v8f zero8() {
  v8f z;
#pragma unroll
  for (int i = 0; i < 8; ++i) z[i] = 0.0f;
  return z;
}

__device__ __forceinline__ v8f wmma_bf16(v16bf a, v16bf b, v8f c) {
  return __builtin_amdgcn_wmma_f32_16x16x32_bf16(false, a, false, b, (short)0, c,
                                                 false, false);
}

// A-matrix 16x32 fragment: lane m=lane&15; lanes 0-15 K=0-15, lanes 16-31 K=16-31.
__device__ __forceinline__ v16bf load_a_frag(const bf16_t* base, int row0, int stride) {
  const int lane = threadIdx.x & 31;
  const int m = lane & 15, hl = lane >> 4;
  const bf16_t* p = base + (size_t)(row0 + m) * stride + hl * 16;
  v8bf lo = *(const v8bf*)(p);
  v8bf hi = *(const v8bf*)(p + 8);
  v16bf f;
#pragma unroll
  for (int i = 0; i < 8; ++i) { f[i] = lo[i]; f[i + 8] = hi[i]; }
  return f;
}

// B-matrix 32x16 fragment from [n][k] storage: lanes 0-15 K=0-15, lanes 16-31 K=16-31.
__device__ __forceinline__ v16bf load_b_frag(const bf16_t* base, int col0, int stride) {
  const int lane = threadIdx.x & 31;
  const int n = lane & 15, hl = lane >> 4;
  const bf16_t* p = base + (size_t)(col0 + n) * stride + hl * 16;
  v8bf lo = *(const v8bf*)(p);
  v8bf hi = *(const v8bf*)(p + 8);
  v16bf f;
#pragma unroll
  for (int i = 0; i < 8; ++i) { f[i] = lo[i]; f[i + 8] = hi[i]; }
  return f;
}

#if USE_TDM
// 2D tensor_load_to_lds per CDNA5 ISA D# layout (groups 0/1; groups 2/3 zero).
// w0 carries data_size=2B + pad_enable/pad_interval/pad_amount bits.
// This toolchain's builtin takes 6 args: (g0 v4u, g1 v8i, v4i, v4i, v8i, cpol).
__device__ __forceinline__ void tdm_load_2d(unsigned lds_addr, const void* gaddr,
                                            unsigned w0, unsigned dim0, unsigned dim1,
                                            unsigned tile0, unsigned tile1,
                                            unsigned stride0) {
  const unsigned long long ga = (unsigned long long)(size_t)gaddr;
  tdm_v4u g0;
  g0[0] = 1u;                                    // count=1 (valid), user mode
  g0[1] = lds_addr;                              // LDS byte address
  g0[2] = (unsigned)ga;                          // global_addr[31:0]
  g0[3] = (unsigned)((ga >> 32) & 0x01FFFFFFu)   // global_addr[56:32]
          | 0x80000000u;                         // type=2 ("image")
  tdm_v8i g1;
  g1[0] = (int)w0;                               // mask=0, data_size, pad ctl
  g1[1] = (int)(dim0 << 16);                     // atomic_barrier_addr=0 | dim0 lo16
  g1[2] = (int)((dim0 >> 16) | (dim1 << 16));    // dim0 hi16 | dim1 lo16
  g1[3] = (int)((dim1 >> 16) | (tile0 << 16));   // dim1 hi16 | tile_dim0
  g1[4] = (int)(tile1 & 0xFFFFu);                // tile_dim1 | tile_dim2=0
  g1[5] = (int)stride0;                          // tensor_dim0_stride lo32
  g1[6] = 0;                                     // stride0 hi16 | stride1 lo16
  g1[7] = 0;
  tdm_v4i z4; z4[0] = z4[1] = z4[2] = z4[3] = 0;
  tdm_v8i z8;
#pragma unroll
  for (int i = 0; i < 8; ++i) z8[i] = 0;
  __builtin_amdgcn_tensor_load_to_lds(g0, g1, z4, z4, z8, 0);
}
#endif

// ---------------------------------------------------------------------------
// Kernel 1: fused Q/K/V projection GEMM, double-buffered LDS stages.
// grid = (24, 64): x = weight*8 + nblk. Q scaled by 1/sqrt(K).
// Q,K written (B,H,T,64); V written transposed (B,H,64,T) for TDM-friendly PV.
// ---------------------------------------------------------------------------
__global__ __launch_bounds__(256)
void qkv_proj_kernel(const float* __restrict__ x,
                     const float* __restrict__ Wq, const float* __restrict__ Wk,
                     const float* __restrict__ Wv,
                     const float* __restrict__ bq, const float* __restrict__ bk,
                     const float* __restrict__ bv,
                     bf16_t* __restrict__ Qb, bf16_t* __restrict__ Kb,
                     bf16_t* __restrict__ Vb) {
  __shared__ alignas(16) bf16_t As[2][128 * AST];   // [row][k]
  __shared__ alignas(16) bf16_t Bs[2][128 * AST];   // [n][k]

  const int wsel = blockIdx.x >> 3;
  const int n0 = (blockIdx.x & 7) * 128;
  const int m0 = blockIdx.y * 128;
  const float* W    = (wsel == 0) ? Wq : ((wsel == 1) ? Wk : Wv);
  const float* bias = (wsel == 0) ? bq : ((wsel == 1) ? bk : bv);
  bf16_t* dst       = (wsel == 0) ? Qb : ((wsel == 1) ? Kb : Vb);
  const float scale = (wsel == 0) ? 0.125f : 1.0f;

  const int tid = threadIdx.x;
  const int wave = tid >> 5;
  const int lane = tid & 31;
  const int hl = lane >> 4;
  const int wm = (wave & 3) * 32;
  const int wn = (wave >> 2) * 64;

  v8f acc[2][4];
#pragma unroll
  for (int i = 0; i < 2; ++i)
#pragma unroll
    for (int j = 0; j < 4; ++j) acc[i][j] = zero8();

  float4 ra[4];          // A prefetch regs
  float4 rb0[2], rb1[2]; // B prefetch regs (two adjacent k-rows for packed stores)

  auto load_tile = [&](int k0) {
#pragma unroll
    for (int it = 0; it < 4; ++it) {
      int idx = tid + it * 256;
      int r = idx >> 3, c = (idx & 7) * 4;
      ra[it] = *(const float4*)(x + (size_t)(m0 + r) * D_ + k0 + c);
    }
#pragma unroll
    for (int it = 0; it < 2; ++it) {
      int idx = tid + it * 256;
      int kp = idx >> 5, nc = (idx & 31) * 4;
      rb0[it] = *(const float4*)(W + (size_t)(k0 + 2 * kp) * HK_ + n0 + nc);
      rb1[it] = *(const float4*)(W + (size_t)(k0 + 2 * kp + 1) * HK_ + n0 + nc);
    }
  };
  auto store_tile = [&](int buf) {
#pragma unroll
    for (int it = 0; it < 4; ++it) {
      int idx = tid + it * 256;
      int r = idx >> 3, c = (idx & 7) * 4;
      union { uint2 u2; __bf16 h[4]; } pk;
      pk.h[0] = (bf16_t)ra[it].x; pk.h[1] = (bf16_t)ra[it].y;
      pk.h[2] = (bf16_t)ra[it].z; pk.h[3] = (bf16_t)ra[it].w;
      *(uint2*)(&As[buf][r * AST + c]) = pk.u2;          // one b64 store
    }
#pragma unroll
    for (int it = 0; it < 2; ++it) {
      int idx = tid + it * 256;
      int kp = idx >> 5, nc = (idx & 31) * 4;
      const float* a = (const float*)&rb0[it];
      const float* b = (const float*)&rb1[it];
#pragma unroll
      for (int q2 = 0; q2 < 4; ++q2) {
        union { unsigned u; __bf16 h[2]; } pk;          // pack (k,k+1) pair
        pk.h[0] = (bf16_t)a[q2]; pk.h[1] = (bf16_t)b[q2];
        *(unsigned*)(&Bs[buf][(nc + q2) * AST + 2 * kp]) = pk.u;
      }
    }
  };

  load_tile(0);
  store_tile(0);
  __syncthreads();

  for (int k0 = 0; k0 < D_; k0 += 32) {
    const int cur = (k0 >> 5) & 1;
    const bool more = (k0 + 32) < D_;
    if (more) load_tile(k0 + 32);                 // global prefetch overlaps WMMA

    v16bf a0 = load_a_frag(As[cur], wm, AST);
    v16bf a1 = load_a_frag(As[cur], wm + 16, AST);
#pragma unroll
    for (int j = 0; j < 4; ++j) {
      v16bf bfr = load_b_frag(Bs[cur], wn + j * 16, AST);
      acc[0][j] = wmma_bf16(a0, bfr, acc[0][j]);
      acc[1][j] = wmma_bf16(a1, bfr, acc[1][j]);
    }
    if (more) store_tile(cur ^ 1);
    __syncthreads();                              // one barrier per stage
  }

#pragma unroll
  for (int j = 0; j < 4; ++j) {
    int nglob = n0 + wn + j * 16 + (lane & 15);
    float bia = bias[nglob];
    int h = nglob >> 6, kf = nglob & 63;
#pragma unroll
    for (int i = 0; i < 2; ++i) {
#pragma unroll
      for (int vg = 0; vg < 8; ++vg) {
        int mglob = m0 + wm + i * 16 + vg + 8 * hl;
        int bb = mglob >> 11, t = mglob & (T_ - 1);
        float y = (acc[i][j][vg] + bia) * scale;
        if (wsel == 2)  // V transposed: (B,H,64,T)
          dst[((size_t)(bb * H_ + h) * K_ + kf) * T_ + t] = (bf16_t)y;
        else            // Q,K: (B,H,T,64)
          dst[((size_t)(bb * H_ + h) * T_ + t) * K_ + kf] = (bf16_t)y;
      }
    }
  }
}

// ---------------------------------------------------------------------------
// Kernel 2: flash attention. grid = (T/256, B*H). Each wave owns 32 query rows.
// K/V chunks double-buffered in LDS, staged by the Tensor Data Mover (wave 0
// issues tensor_load_to_lds one chunk ahead; s_wait_tensorcnt + barrier gates).
// ---------------------------------------------------------------------------
__global__ __launch_bounds__(256)
void attn_kernel(const bf16_t* __restrict__ Qb, const bf16_t* __restrict__ Kb,
                 const bf16_t* __restrict__ Vb, bf16_t* __restrict__ Ctx) {
  __shared__ alignas(16) bf16_t Ks2[2][32 * KST];   // [kv][feat], stride 72 (TDM pad)
  __shared__ alignas(16) bf16_t Vt2[2][64 * AST];   // [feat][kv], stride 40 (TDM pad)
  __shared__ alignas(16) bf16_t Ps[8 * 32 * AST];   // per-wave P scratch

  const int bh = blockIdx.y;
  const int b = bh >> 4, h = bh & 15;
  const int t0 = blockIdx.x * 256;
  const bf16_t* Qp = Qb + (size_t)bh * T_ * K_;
  const bf16_t* Kp = Kb + (size_t)bh * T_ * K_;
  const bf16_t* Vp = Vb + (size_t)bh * K_ * T_;     // transposed (feat, t)

  const int tid = threadIdx.x;
  const int wave = tid >> 5;
  const int lane = tid & 31;
  const int hl = lane >> 4;
  const int qrow0 = t0 + wave * 32;

  v16bf qa[2][2];
#pragma unroll
  for (int i = 0; i < 2; ++i)
#pragma unroll
    for (int kk = 0; kk < 2; ++kk)
      qa[i][kk] = load_a_frag(Qp + kk * 32, qrow0 + i * 16, K_);

  v8f o[2][4];
#pragma unroll
  for (int i = 0; i < 2; ++i)
#pragma unroll
    for (int j = 0; j < 4; ++j) o[i][j] = zero8();

  float row_max[16], row_sum[16];
#pragma unroll
  for (int r = 0; r < 16; ++r) { row_max[r] = -1e30f; row_sum[r] = 0.0f; }

  bf16_t* pw = Ps + wave * 32 * AST;

  auto issue_chunk = [&](int c0, int buf) {
#if USE_TDM
    if (wave == 0) {
      // K chunk: 32 rows x 64 feats, row-contiguous; LDS padded 32dw+4dw -> stride 72.
      tdm_load_2d((unsigned)(size_t)&Ks2[buf][0], Kp + (size_t)c0 * K_,
                  /*w0: ds=2B,pad,intv=32dw,amt=4dw*/ 0x07110000u,
                  /*dim0*/ 64u, /*dim1*/ 2048u, /*tile0*/ 64u, /*tile1*/ 32u,
                  /*stride0*/ 64u);
      // V chunk: 64 feat rows x 32 t cols from (feat,t); LDS 16dw+4dw -> stride 40.
      tdm_load_2d((unsigned)(size_t)&Vt2[buf][0], Vp + c0,
                  /*w0: ds=2B,pad,intv=16dw,amt=4dw*/ 0x06D10000u,
                  /*dim0*/ 2048u, /*dim1*/ 64u, /*tile0*/ 32u, /*tile1*/ 64u,
                  /*stride0*/ 2048u);
    }
#else
    {
      int kv = tid >> 3, f8 = (tid & 7) * 8;
      *(uint4*)(&Ks2[buf][kv * KST + f8]) =
          *(const uint4*)(Kp + (size_t)(c0 + kv) * K_ + f8);
      int f = tid >> 2, g = (tid & 3) * 8;
      *(uint4*)(&Vt2[buf][f * AST + g]) =
          *(const uint4*)(Vp + (size_t)f * T_ + c0 + g);
    }
#endif
  };

  issue_chunk(0, 0);

  for (int c0 = 0, it = 0; c0 < T_; c0 += 32, ++it) {
    const int buf = it & 1;
#if USE_TDM
    if (wave == 0) __builtin_amdgcn_s_wait_tensorcnt(0);
#endif
    __syncthreads();                               // chunk `it` landed in LDS
    if (c0 + 32 < T_) issue_chunk(c0 + 32, buf ^ 1);

    // S = Q * K^T (32x32 per wave; 2 chained WMMAs over 64 feats)
    v8f s[2][2];
#pragma unroll
    for (int j = 0; j < 2; ++j) {
      v16bf kb0 = load_b_frag(Ks2[buf],      j * 16, KST);
      v16bf kb1 = load_b_frag(Ks2[buf] + 32, j * 16, KST);
#pragma unroll
      for (int i = 0; i < 2; ++i) {
        v8f tacc = wmma_bf16(qa[i][0], kb0, zero8());
        s[i][j] = wmma_bf16(qa[i][1], kb1, tacc);
      }
    }

    // Online softmax: rows live across 16-lane halves.
#pragma unroll
    for (int i = 0; i < 2; ++i) {
#pragma unroll
      for (int vg = 0; vg < 8; ++vg) {
        float mx = fmaxf(s[i][0][vg], s[i][1][vg]);
        mx = fmaxf(mx, __shfl_xor(mx, 8, 16));
        mx = fmaxf(mx, __shfl_xor(mx, 4, 16));
        mx = fmaxf(mx, __shfl_xor(mx, 2, 16));
        mx = fmaxf(mx, __shfl_xor(mx, 1, 16));
        const int r = i * 8 + vg;
        float nm = fmaxf(row_max[r], mx);
        float corr = __expf(row_max[r] - nm);
        row_max[r] = nm;
        float p0 = __expf(s[i][0][vg] - nm);
        float p1 = __expf(s[i][1][vg] - nm);
        s[i][0][vg] = p0; s[i][1][vg] = p1;
        float ps = p0 + p1;
        ps += __shfl_xor(ps, 8, 16);
        ps += __shfl_xor(ps, 4, 16);
        ps += __shfl_xor(ps, 2, 16);
        ps += __shfl_xor(ps, 1, 16);
        row_sum[r] = row_sum[r] * corr + ps;
#pragma unroll
        for (int j = 0; j < 4; ++j) o[i][j][vg] *= corr;
      }
    }

    // P (D-layout) -> LDS -> A-layout fragments, then O += P * V.
#pragma unroll
    for (int i = 0; i < 2; ++i)
#pragma unroll
      for (int j = 0; j < 2; ++j)
#pragma unroll
        for (int vg = 0; vg < 8; ++vg)
          pw[(i * 16 + vg + 8 * hl) * AST + j * 16 + (lane & 15)] = (bf16_t)s[i][j][vg];
    __syncthreads();

    v16bf pa0 = load_a_frag(pw, 0, AST);
    v16bf pa1 = load_a_frag(pw, 16, AST);
#pragma unroll
    for (int j = 0; j < 4; ++j) {
      v16bf vb = load_b_frag(Vt2[buf], j * 16, AST);
      o[0][j] = wmma_bf16(pa0, vb, o[0][j]);
      o[1][j] = wmma_bf16(pa1, vb, o[1][j]);
    }
  }

  // Normalize and write context to (B,T,H,64) = (M, 1024) bf16.
  const int n = lane & 15;
#pragma unroll
  for (int i = 0; i < 2; ++i) {
#pragma unroll
    for (int vg = 0; vg < 8; ++vg) {
      const int r = i * 8 + vg;
      float inv = 1.0f / row_sum[r];
      int t = qrow0 + i * 16 + vg + 8 * hl;
#pragma unroll
      for (int j = 0; j < 4; ++j) {
        int feat = j * 16 + n;
        Ctx[(size_t)(b * T_ + t) * HK_ + h * K_ + feat] = (bf16_t)(o[i][j][vg] * inv);
      }
    }
  }
}

// ---------------------------------------------------------------------------
// Kernel 3: output projection, double-buffered. grid = (8, 64).
// out = Ctx @ W_o + b_o (fp32 out).
// ---------------------------------------------------------------------------
__global__ __launch_bounds__(256)
void out_proj_kernel(const bf16_t* __restrict__ Ctx, const float* __restrict__ Wo,
                     const float* __restrict__ bo, float* __restrict__ out) {
  __shared__ alignas(16) bf16_t As[2][128 * AST];
  __shared__ alignas(16) bf16_t Bs[2][128 * AST];
  const int n0 = blockIdx.x * 128;
  const int m0 = blockIdx.y * 128;
  const int tid = threadIdx.x;
  const int wave = tid >> 5, lane = tid & 31, hl = lane >> 4;
  const int wm = (wave & 3) * 32, wn = (wave >> 2) * 64;

  v8f acc[2][4];
#pragma unroll
  for (int i = 0; i < 2; ++i)
#pragma unroll
    for (int j = 0; j < 4; ++j) acc[i][j] = zero8();

  uint4 rca[2];
  float4 rb0[2], rb1[2];

  auto load_tile = [&](int k0) {
#pragma unroll
    for (int w = 0; w < 2; ++w) {   // A: bf16 direct (128x32)
      int idx = tid + w * 256;
      int r = idx >> 2, c8 = (idx & 3) * 8;
      rca[w] = *(const uint4*)(Ctx + (size_t)(m0 + r) * HK_ + k0 + c8);
    }
#pragma unroll
    for (int it = 0; it < 2; ++it) { // B: two adjacent k-rows of Wo
      int idx = tid + it * 256;
      int kp = idx >> 5, nc = (idx & 31) * 4;
      rb0[it] = *(const float4*)(Wo + (size_t)(k0 + 2 * kp) * D_ + n0 + nc);
      rb1[it] = *(const float4*)(Wo + (size_t)(k0 + 2 * kp + 1) * D_ + n0 + nc);
    }
  };
  auto store_tile = [&](int buf) {
#pragma unroll
    for (int w = 0; w < 2; ++w) {
      int idx = tid + w * 256;
      int r = idx >> 2, c8 = (idx & 3) * 8;
      *(uint4*)(&As[buf][r * AST + c8]) = rca[w];
    }
#pragma unroll
    for (int it = 0; it < 2; ++it) {
      int idx = tid + it * 256;
      int kp = idx >> 5, nc = (idx & 31) * 4;
      const float* a = (const float*)&rb0[it];
      const float* b = (const float*)&rb1[it];
#pragma unroll
      for (int q2 = 0; q2 < 4; ++q2) {
        union { unsigned u; __bf16 h[2]; } pk;
        pk.h[0] = (bf16_t)a[q2]; pk.h[1] = (bf16_t)b[q2];
        *(unsigned*)(&Bs[buf][(nc + q2) * AST + 2 * kp]) = pk.u;
      }
    }
  };

  load_tile(0);
  store_tile(0);
  __syncthreads();

  for (int k0 = 0; k0 < HK_; k0 += 32) {
    const int cur = (k0 >> 5) & 1;
    const bool more = (k0 + 32) < HK_;
    if (more) load_tile(k0 + 32);

    v16bf a0 = load_a_frag(As[cur], wm, AST);
    v16bf a1 = load_a_frag(As[cur], wm + 16, AST);
#pragma unroll
    for (int j = 0; j < 4; ++j) {
      v16bf bfr = load_b_frag(Bs[cur], wn + j * 16, AST);
      acc[0][j] = wmma_bf16(a0, bfr, acc[0][j]);
      acc[1][j] = wmma_bf16(a1, bfr, acc[1][j]);
    }
    if (more) store_tile(cur ^ 1);
    __syncthreads();
  }

#pragma unroll
  for (int j = 0; j < 4; ++j) {
    int nglob = n0 + wn + j * 16 + (lane & 15);
    float bia = bo[nglob];
#pragma unroll
    for (int i = 0; i < 2; ++i) {
#pragma unroll
      for (int vg = 0; vg < 8; ++vg) {
        int mglob = m0 + wm + i * 16 + vg + 8 * hl;
        out[(size_t)mglob * D_ + nglob] = acc[i][j][vg] + bia;
      }
    }
  }
}

// ---------------------------------------------------------------------------
extern "C" void kernel_launch(void* const* d_in, const int* in_sizes, int n_in,
                              void* d_out, int out_size, void* d_ws, size_t ws_size,
                              hipStream_t stream) {
  const float* q  = (const float*)d_in[0];
  const float* Wq = (const float*)d_in[1];
  const float* Wk = (const float*)d_in[2];
  const float* Wv = (const float*)d_in[3];
  const float* bq = (const float*)d_in[4];
  const float* bk = (const float*)d_in[5];
  const float* bv = (const float*)d_in[6];
  const float* Wo = (const float*)d_in[7];
  const float* bo = (const float*)d_in[8];
  float* out = (float*)d_out;

  // Workspace: 4 bf16 buffers of M_*HK_ elements = 64 MB total.
  const size_t SZ = (size_t)M_ * HK_;
  bf16_t* Qb  = (bf16_t*)d_ws;
  bf16_t* Kb  = Qb + SZ;
  bf16_t* Vb  = Kb + SZ;     // stored transposed per (b,h): (64, T)
  bf16_t* Ctx = Vb + SZ;

  qkv_proj_kernel<<<dim3(24, 64), 256, 0, stream>>>(q, Wq, Wk, Wv, bq, bk, bv,
                                                    Qb, Kb, Vb);
  attn_kernel<<<dim3(T_ / 256, B_ * H_), 256, 0, stream>>>(Qb, Kb, Vb, Ctx);
  out_proj_kernel<<<dim3(8, 64), 256, 0, stream>>>(Ctx, Wo, bo, out);
}
